// OurDynamics_90726889161252
// MI455X (gfx1250) — matrix operand order repaired and checked by reading, not compiled, not run on hardware
//
#include <hip/hip_runtime.h>
#include <hip/hip_bf16.h>
#include <math.h>

#define DEVINL __device__ __forceinline__

typedef __attribute__((ext_vector_type(16))) __bf16 v16bf;
typedef __attribute__((ext_vector_type(8)))  __bf16 v8bf;
typedef __attribute__((ext_vector_type(8)))  float  v8f;

union bf16x16 { v16bf v; v8bf h[2]; };

DEVINL v8f wmma_bf16(v16bf a, v16bf b, v8f c) {
  // D = A(16x32 bf16) * B(32x16 bf16) + C(16x16 f32)
  return __builtin_amdgcn_wmma_f32_16x16x32_bf16(
      /*neg_a=*/false, a, /*neg_b=*/false, b,
      /*c_mod=*/(short)0, c, /*reuse_a=*/false, /*reuse_b=*/false);
}

DEVINL v8bf zero_v8bf() {
  v8bf z;
#pragma unroll
  for (int i = 0; i < 8; ++i) z[i] = (__bf16)0.f;
  return z;
}

// Wave-local LDS ordering (cross-lane exchange within one wave32).
DEVINL void lds_wave_sync() {
  __builtin_amdgcn_wave_barrier();
  asm volatile("s_wait_dscnt 0" ::: "memory");
  __builtin_amdgcn_wave_barrier();
}

// A operand 16x32 bf16 (MxK): lane's two 8-element contiguous runs.
// kbase = k0 + ((lane>>4)<<3); rowp = row base for this lane's M.
// Requires 16B alignment of rowp+kbase (rows are 16*2B-aligned, kbase%8==0).
DEVINL v16bf load_a_bf(const __bf16* rowp, int kbase) {
  bf16x16 u;
  u.h[0] = *(const v8bf*)(rowp + kbase);
  u.h[1] = *(const v8bf*)(rowp + kbase + 16);
  return u.v;
}
// B operand 32x16 bf16: pre-swizzled so each lane's 16 values are contiguous.
DEVINL v16bf load_b_swz(const __bf16* p) {
  bf16x16 u;
  u.h[0] = *(const v8bf*)(p);
  u.h[1] = *(const v8bf*)(p + 8);
  return u.v;
}

// ===========================================================================
// Prep: swizzle f32 weight [K][ldn] into bf16 B-operand layout
//   dst[((nt*nChunks + c)*32 + lane)*16 + i]
//     = bf16( W[(c*32 + ((lane>>4)<<3) + (i&7) + ((i>>3)<<4)) * ldn + nt*16 + (lane&15)] )
// K zero-padded to nChunks*32.
// ===========================================================================
__global__ void swizzle_b_kernel(const float* __restrict__ W, int K, int ldn,
                                 int nChunks, int nTilesN, __bf16* __restrict__ dst) {
  const int total = nTilesN * nChunks * 32 * 16;
  for (int t = blockIdx.x * blockDim.x + threadIdx.x; t < total;
       t += gridDim.x * blockDim.x) {
    const int i = t & 15;
    const int lane = (t >> 4) & 31;
    const int rem = t >> 9;
    const int c = rem % nChunks;
    const int nt = rem / nChunks;
    const int k = c * 32 + ((lane >> 4) << 3) + (i & 7) + ((i >> 3) << 4);
    const int n = nt * 16 + (lane & 15);
    dst[t] = (k < K) ? (__bf16)W[(size_t)k * ldn + n] : (__bf16)0.f;
  }
}

__global__ void f32_to_bf16_kernel(const float* __restrict__ src,
                                   __bf16* __restrict__ dst, int n) {
  int t = blockIdx.x * blockDim.x + threadIdx.x;
  if (t < n) dst[t] = (__bf16)src[t];
}
__global__ void f32_to_bf16_relu_kernel(const float* __restrict__ src,
                                        __bf16* __restrict__ dst, int n) {
  int t = blockIdx.x * blockDim.x + threadIdx.x;
  if (t < n) dst[t] = (__bf16)fmaxf(src[t], 0.f);
}

// ===========================================================================
// Kernel 1: coordinate embedding MLP  x = MLP(relu(nodes))  [N,256] -> [N,3]
// One wave = one 16-node tile; 4 waves per block. Activations bf16 in LDS.
// ===========================================================================
__global__ void embed_kernel(const __bf16* __restrict__ nrelu,
                             const __bf16* __restrict__ W0s, const float* __restrict__ b0,
                             const __bf16* __restrict__ W1s, const float* __restrict__ b1,
                             const __bf16* __restrict__ W2s, const float* __restrict__ b2,
                             const float* __restrict__ W3, const float* __restrict__ b3,
                             float* __restrict__ x, int N) {
  __shared__ __attribute__((aligned(16))) __bf16 h1s[4][16 * 128];
  __shared__ __attribute__((aligned(16))) __bf16 h2s[4][16 * 64];
  __shared__ __attribute__((aligned(16))) __bf16 h3s[4][16 * 32];
  const int lane = threadIdx.x & 31;
  const int wave = threadIdx.x >> 5;
  const int row0 = (blockIdx.x * 4 + wave) * 16;
  if (row0 >= N) return;
  const int n15 = lane & 15;
  const int hi8 = (lane >> 4) << 3;
  __bf16* h1 = h1s[wave];
  __bf16* h2 = h2s[wave];
  __bf16* h3 = h3s[wave];

  // layer 1: relu(nodes)[16,256] @ W0[256,128], relu
  {
    const __bf16* arow = nrelu + (size_t)(row0 + n15) * 256;
#pragma unroll
    for (int nt = 0; nt < 8; ++nt) {
      const int col = nt * 16 + n15;
      v8f acc = {};
      for (int kt = 0; kt < 8; ++kt) {
        v16bf a = load_a_bf(arow, kt * 32 + hi8);
        v16bf b = load_b_swz(W0s + ((nt * 8 + kt) * 32 + lane) * 16);
        acc = wmma_bf16(a, b, acc);
      }
      const float bias = b0[col];
#pragma unroll
      for (int r = 0; r < 8; ++r)
        h1[(r + hi8) * 128 + col] = (__bf16)fmaxf(acc[r] + bias, 0.f);
    }
  }
  lds_wave_sync();
  // layer 2: h1[16,128] @ W1[128,64], relu
  {
    const __bf16* arow = h1 + n15 * 128;
#pragma unroll
    for (int nt = 0; nt < 4; ++nt) {
      const int col = nt * 16 + n15;
      v8f acc = {};
      for (int kt = 0; kt < 4; ++kt) {
        v16bf a = load_a_bf(arow, kt * 32 + hi8);
        v16bf b = load_b_swz(W1s + ((nt * 4 + kt) * 32 + lane) * 16);
        acc = wmma_bf16(a, b, acc);
      }
      const float bias = b1[col];
#pragma unroll
      for (int r = 0; r < 8; ++r)
        h2[(r + hi8) * 64 + col] = (__bf16)fmaxf(acc[r] + bias, 0.f);
    }
  }
  lds_wave_sync();
  // layer 3: h2[16,64] @ W2[64,32], relu
  {
    const __bf16* arow = h2 + n15 * 64;
#pragma unroll
    for (int nt = 0; nt < 2; ++nt) {
      const int col = nt * 16 + n15;
      v8f acc = {};
      for (int kt = 0; kt < 2; ++kt) {
        v16bf a = load_a_bf(arow, kt * 32 + hi8);
        v16bf b = load_b_swz(W2s + ((nt * 2 + kt) * 32 + lane) * 16);
        acc = wmma_bf16(a, b, acc);
      }
      const float bias = b2[col];
#pragma unroll
      for (int r = 0; r < 8; ++r)
        h3[(r + hi8) * 32 + col] = (__bf16)fmaxf(acc[r] + bias, 0.f);
    }
  }
  lds_wave_sync();
  // layer 4 (tiny): h3[16,32] @ W3[32,3] + b3, scalar on lanes 0..15
  if (lane < 16) {
    const int m = lane;
#pragma unroll
    for (int j = 0; j < 3; ++j) {
      float s = b3[j];
      for (int k = 0; k < 32; ++k) s += (float)h3[m * 32 + k] * W3[k * 3 + j];
      x[(size_t)(row0 + m) * 3 + j] = s;
    }
  }
}

// ===========================================================================
// Kernel 2: in-degree
// ===========================================================================
__global__ void deg_kernel(const int* __restrict__ edges, float* __restrict__ deg, int E) {
  int t = blockIdx.x * blockDim.x + threadIdx.x;
  if (t < E) atomicAdd(&deg[edges[E + t]], 1.0f);
}

// ===========================================================================
// Kernel 3: per-layer edge MLP + scatter-add aggregation.
// h in bf16 (A = two b128 loads); We1/We2 pre-swizzled bf16 staged in LDS.
// ===========================================================================
__global__ void edge_kernel(const __bf16* __restrict__ hbf, const float* __restrict__ x,
                            const int* __restrict__ edges, const float* __restrict__ edge_attr,
                            const __bf16* __restrict__ We1s, const float* __restrict__ be1,
                            const __bf16* __restrict__ We2s, const float* __restrict__ be2,
                            const float* __restrict__ Wc,  const float* __restrict__ bc,
                            float* __restrict__ agg_x, float* __restrict__ agg_m, int E) {
  __shared__ __attribute__((aligned(16))) __bf16 W1s[17 * 32 * 16]; // K padded to 544
  __shared__ __attribute__((aligned(16))) __bf16 W2s[32 * 16];      // K padded to 32
  __shared__ __attribute__((aligned(16))) __bf16 ms[8][16 * 16];

  {  // stage swizzled weights (raw u32 copy, already bf16)
    const uint32_t* s1 = (const uint32_t*)We1s;
    uint32_t* d1 = (uint32_t*)W1s;
    for (int i = threadIdx.x; i < 17 * 512 / 2; i += blockDim.x) d1[i] = s1[i];
    const uint32_t* s2 = (const uint32_t*)We2s;
    uint32_t* d2 = (uint32_t*)W2s;
    for (int i = threadIdx.x; i < 512 / 2; i += blockDim.x) d2[i] = s2[i];
  }
  __syncthreads();

  const int lane = threadIdx.x & 31;
  const int wave = threadIdx.x >> 5;
  const int n15 = lane & 15;
  const int hi8 = (lane >> 4) << 3;
  __bf16* msw = ms[wave];
  const float b1v = be1[n15];
  const float b2v = be2[n15];

  const int nTiles = E >> 4;
  const int waveGlobal = blockIdx.x * (blockDim.x >> 5) + wave;
  const int nWaves = gridDim.x * (blockDim.x >> 5);

  for (int tile = waveGlobal; tile < nTiles; tile += nWaves) {
    const int row0 = tile << 4;
    const int e = row0 + n15;
    const int srcI = edges[e];
    const int dstI = edges[E + e];
    const __bf16* hs = hbf + (size_t)srcI * 256;
    const __bf16* hd = hbf + (size_t)dstI * 256;
    const float rx = x[srcI * 3 + 0] - x[dstI * 3 + 0];
    const float ry = x[srcI * 3 + 1] - x[dstI * 3 + 1];
    const float rz = x[srcI * 3 + 2] - x[dstI * 3 + 2];
    const float r2 = rx * rx + ry * ry + rz * rz;
    const float ea = edge_attr[e];

    // m1 = relu(feat @ We1 + be1):  17 K-chunks of 32
    v8f acc = {};
    for (int kt = 0; kt < 8; ++kt) {                     // h[src] -> cols 0..255
      v16bf a = load_a_bf(hs, kt * 32 + hi8);
      v16bf b = load_b_swz(W1s + (kt * 32 + lane) * 16);
      acc = wmma_bf16(a, b, acc);
    }
    for (int kt = 0; kt < 8; ++kt) {                     // h[dst] -> cols 256..511
      v16bf a = load_a_bf(hd, kt * 32 + hi8);
      v16bf b = load_b_swz(W1s + ((8 + kt) * 32 + lane) * 16);
      acc = wmma_bf16(a, b, acc);
    }
    {                                                    // [r2, ea] -> cols 512..513
      bf16x16 ua;
      ua.h[0] = zero_v8bf();
      ua.h[1] = zero_v8bf();
      if (lane < 16) { ua.h[0][0] = (__bf16)r2; ua.h[0][1] = (__bf16)ea; }
      v16bf b = load_b_swz(W1s + (16 * 32 + lane) * 16); // rows >=514 zero-padded
      acc = wmma_bf16(ua.v, b, acc);
    }
#pragma unroll
    for (int r = 0; r < 8; ++r)
      msw[(r + hi8) * 16 + n15] = (__bf16)fmaxf(acc[r] + b1v, 0.f);
    lds_wave_sync();

    // m2 = relu(m1 @ We2 + be2): single WMMA, K padded 16 -> 32
    bf16x16 ua2;
    ua2.h[0] = *(const v8bf*)(msw + n15 * 16 + hi8);   // valid K (cols 0..15)
    ua2.h[1] = zero_v8bf();                            // pad K 16..31
    v16bf b2 = load_b_swz(W2s + lane * 16);
    v8f acc2 = wmma_bf16(ua2.v, b2, (v8f){});
    lds_wave_sync();
#pragma unroll
    for (int r = 0; r < 8; ++r)
      msw[(r + hi8) * 16 + n15] = (__bf16)fmaxf(acc2[r] + b2v, 0.f);
    lds_wave_sync();

    // coef = m2 @ Wc + bc, then agg_x[dst] += coef * rel   (lanes 0..15)
    if (lane < 16) {
      float cf = bc[0];
#pragma unroll
      for (int k = 0; k < 16; ++k) cf += (float)msw[n15 * 16 + k] * Wc[k];
      atomicAdd(&agg_x[dstI * 3 + 0], cf * rx);
      atomicAdd(&agg_x[dstI * 3 + 1], cf * ry);
      atomicAdd(&agg_x[dstI * 3 + 2], cf * rz);
    }
    // agg_m[dst] += m2  (all lanes, from accumulator registers)
#pragma unroll
    for (int r = 0; r < 8; ++r) {
      const int M = r + hi8;
      const int d = edges[E + row0 + M];
      atomicAdd(&agg_m[d * 16 + n15], fmaxf(acc2[r] + b2v, 0.f));
    }
  }
}

// ===========================================================================
// Kernel 4: node update.  x += agg_x/deg;  h += relu([h|agg_m] @ Wh + bh)
// Wh pre-swizzled bf16 (K padded 272 -> 288, 9 chunks, 16 n-tiles).
// Writes both f32 h_out (residual master) and bf16 mirror.
// ===========================================================================
__global__ void node_update_kernel(const float* __restrict__ h_in,
                                   const __bf16* __restrict__ h_in_bf,
                                   const float* __restrict__ x_in,
                                   const float* __restrict__ agg_x, const float* __restrict__ agg_m,
                                   const float* __restrict__ deg,
                                   const __bf16* __restrict__ Whs, const float* __restrict__ bh,
                                   float* __restrict__ h_out, __bf16* __restrict__ h_out_bf,
                                   float* __restrict__ x_out, int N) {
  const int lane = threadIdx.x & 31;
  const int wave = threadIdx.x >> 5;
  const int row0 = (blockIdx.x * (blockDim.x >> 5) + wave) * 16;
  if (row0 >= N) return;
  const int n15 = lane & 15;
  const int hi8 = (lane >> 4) << 3;
  const __bf16* hrow = h_in_bf + (size_t)(row0 + n15) * 256;
  const float* arow = agg_m + (size_t)(row0 + n15) * 16;

  if (lane < 16) {
    const int i = row0 + n15;
    const float d = fmaxf(deg[i], 1.f);
#pragma unroll
    for (int j = 0; j < 3; ++j)
      x_out[i * 3 + j] = x_in[i * 3 + j] + agg_x[i * 3 + j] / d;
  }

  // A chunk 8 (cols 256..287): agg_m in 256..271, zero pad above.
  bf16x16 ua8;
#pragma unroll
  for (int i = 0; i < 8; ++i) ua8.h[0][i] = (__bf16)arow[hi8 + i];
  ua8.h[1] = zero_v8bf();

  for (int nt = 0; nt < 16; ++nt) {
    const int col = nt * 16 + n15;
    v8f acc = {};
    for (int kt = 0; kt < 8; ++kt) {                     // h part, K = 0..255
      v16bf a = load_a_bf(hrow, kt * 32 + hi8);
      v16bf b = load_b_swz(Whs + ((nt * 9 + kt) * 32 + lane) * 16);
      acc = wmma_bf16(a, b, acc);
    }
    {                                                    // agg_m part, K = 256..287
      v16bf b = load_b_swz(Whs + ((nt * 9 + 8) * 32 + lane) * 16);
      acc = wmma_bf16(ua8.v, b, acc);
    }
    const float bias = bh[col];
#pragma unroll
    for (int r = 0; r < 8; ++r) {
      const size_t idx = (size_t)(row0 + r + hi8) * 256 + col;
      const float v = h_in[idx] + fmaxf(acc[r] + bias, 0.f);
      h_out[idx] = v;
      h_out_bf[idx] = (__bf16)v;
    }
  }
}

// ===========================================================================
// Kernel 5: outputs.  d_out = [x | x | cdist(x,x)]
// ===========================================================================
__global__ void copy_out_kernel(const float* __restrict__ xf, float* __restrict__ out, int n3) {
  int t = blockIdx.x * blockDim.x + threadIdx.x;
  if (t < n3) { out[t] = xf[t]; out[n3 + t] = xf[t]; }
}

__global__ void cdist_kernel(const float* __restrict__ xf, float* __restrict__ dist, int N) {
  const int i = blockIdx.x;
  const float xi0 = xf[i * 3 + 0], xi1 = xf[i * 3 + 1], xi2 = xf[i * 3 + 2];
  for (int j = threadIdx.x; j < N; j += blockDim.x) {
    const float d0 = xi0 - xf[j * 3 + 0];
    const float d1 = xi1 - xf[j * 3 + 1];
    const float d2c = xi2 - xf[j * 3 + 2];
    const float d2 = d0 * d0 + d1 * d1 + d2c * d2c;
    dist[(size_t)i * N + j] = (d2 > 1e-9f) ? sqrtf(d2) : 0.f;
  }
}

// ===========================================================================
extern "C" void kernel_launch(void* const* d_in, const int* in_sizes, int n_in,
                              void* d_out, int out_size, void* d_ws, size_t ws_size,
                              hipStream_t stream) {
  const float* nodes     = (const float*)d_in[0];
  const int*   edges     = (const int*)d_in[1];
  const float* edge_attr = (const float*)d_in[2];
  const float* W0 = (const float*)d_in[3];   const float* b0 = (const float*)d_in[4];
  const float* W1 = (const float*)d_in[5];   const float* b1 = (const float*)d_in[6];
  const float* W2 = (const float*)d_in[7];   const float* b2 = (const float*)d_in[8];
  const float* W3 = (const float*)d_in[9];   const float* b3 = (const float*)d_in[10];
  const float* We1 = (const float*)d_in[11]; const float* be1 = (const float*)d_in[12];
  const float* We2 = (const float*)d_in[13]; const float* be2 = (const float*)d_in[14];
  const float* Wc  = (const float*)d_in[15]; const float* bc  = (const float*)d_in[16];
  const float* Wh  = (const float*)d_in[17]; const float* bh  = (const float*)d_in[18];

  const int N = in_sizes[0] / 256;   // 8192
  const int E = in_sizes[1] / 2;     // 262144

  // Workspace carve-out
  char* w = (char*)d_ws;
  auto carve = [&](size_t bytes) -> char* {
    char* p = w;
    w += (bytes + 255) & ~(size_t)255;
    return p;
  };
  float*  xA     = (float*)carve((size_t)N * 3 * 4);
  float*  xB     = (float*)carve((size_t)N * 3 * 4);
  float*  hA     = (float*)carve((size_t)N * 256 * 4);
  float*  hB     = (float*)carve((size_t)N * 256 * 4);
  __bf16* hAbf   = (__bf16*)carve((size_t)N * 256 * 2);
  __bf16* hBbf   = (__bf16*)carve((size_t)N * 256 * 2);
  __bf16* nbf    = (__bf16*)carve((size_t)N * 256 * 2);
  __bf16* nrbf   = (__bf16*)carve((size_t)N * 256 * 2);
  float*  aggx   = (float*)carve((size_t)N * 3 * 4);
  float*  aggm   = (float*)carve((size_t)N * 16 * 4);
  float*  deg    = (float*)carve((size_t)N * 4);
  __bf16* W0s    = (__bf16*)carve(8 * 8 * 512 * 2);
  __bf16* W1sz   = (__bf16*)carve(4 * 4 * 512 * 2);
  __bf16* W2sz   = (__bf16*)carve(2 * 2 * 512 * 2);
  __bf16* We1s   = (__bf16*)carve(3 * 17 * 512 * 2);
  __bf16* We2s   = (__bf16*)carve(3 * 1 * 512 * 2);
  __bf16* Whs    = (__bf16*)carve(3 * 16 * 9 * 512 * 2);

  // ---- one-time format prep (bf16 mirrors + B-operand swizzles) ----
  const int nEl = N * 256;
  f32_to_bf16_kernel<<<(nEl + 255) / 256, 256, 0, stream>>>(nodes, nbf, nEl);
  f32_to_bf16_relu_kernel<<<(nEl + 255) / 256, 256, 0, stream>>>(nodes, nrbf, nEl);
  swizzle_b_kernel<<<128, 256, 0, stream>>>(W0, 256, 128, 8, 8, W0s);
  swizzle_b_kernel<<<32, 256, 0, stream>>>(W1, 128, 64, 4, 4, W1sz);
  swizzle_b_kernel<<<8, 256, 0, stream>>>(W2, 64, 32, 2, 2, W2sz);
  for (int l = 0; l < 3; ++l) {
    swizzle_b_kernel<<<34, 256, 0, stream>>>(We1 + (size_t)l * 514 * 16, 514, 16, 17, 1,
                                             We1s + (size_t)l * 17 * 512);
    swizzle_b_kernel<<<2, 256, 0, stream>>>(We2 + (size_t)l * 256, 16, 16, 1, 1,
                                            We2s + (size_t)l * 512);
    swizzle_b_kernel<<<288, 256, 0, stream>>>(Wh + (size_t)l * 272 * 256, 272, 256, 9, 16,
                                              Whs + (size_t)l * 16 * 9 * 512);
  }

  embed_kernel<<<N / 64, 128, 0, stream>>>(nrbf, W0s, b0, W1sz, b1, W2sz, b2, W3, b3, xA, N);

  hipMemsetAsync(deg, 0, (size_t)N * 4, stream);
  deg_kernel<<<(E + 255) / 256, 256, 0, stream>>>(edges, deg, E);

  const float*  h_cur    = nodes;
  const __bf16* h_cur_bf = nbf;
  const float*  x_cur    = xA;
  float*  hbuf[2]  = {hA, hB};
  __bf16* hbufb[2] = {hAbf, hBbf};
  float*  xbuf[2]  = {xB, xA};  // layer0 -> xB, layer1 -> xA, layer2 -> xB
  for (int i = 0; i < 3; ++i) {
    hipMemsetAsync(aggx, 0, (size_t)N * 3 * 4, stream);
    hipMemsetAsync(aggm, 0, (size_t)N * 16 * 4, stream);
    edge_kernel<<<256, 256, 0, stream>>>(h_cur_bf, x_cur, edges, edge_attr,
        We1s + (size_t)i * 17 * 512, be1 + i * 16,
        We2s + (size_t)i * 512,      be2 + i * 16,
        Wc + i * 16, bc + i, aggx, aggm, E);
    float*  h_nxt  = hbuf[i & 1];
    __bf16* h_nxtb = hbufb[i & 1];
    float*  x_nxt  = xbuf[i & 1];
    node_update_kernel<<<N / 128, 256, 0, stream>>>(h_cur, h_cur_bf, x_cur, aggx, aggm, deg,
        Whs + (size_t)i * 16 * 9 * 512, bh + i * 256, h_nxt, h_nxtb, x_nxt, N);
    h_cur = h_nxt;
    h_cur_bf = h_nxtb;
    x_cur = x_nxt;
  }

  float* out = (float*)d_out;
  copy_out_kernel<<<(N * 3 + 255) / 256, 256, 0, stream>>>(x_cur, out, N * 3);
  cdist_kernel<<<N, 256, 0, stream>>>(x_cur, out + 2 * (size_t)N * 3, N);
}
